// AttentionModule_16432544875314
// MI455X (gfx1250) — matrix operand order repaired and checked by reading, not compiled, not run on hardware
//
#include <hip/hip_runtime.h>
#include <hip/hip_bf16.h>
#include <math.h>

// ---------------- problem constants (from reference) ----------------
#define S_    250
#define N_    200
#define MAXP  16
#define H_    64
#define E_    64
#define HID_  512
#define BN_   32
#define TOK   (S_ * N_)          // 50000 tokens

typedef __attribute__((ext_vector_type(16))) _Float16 v16h;
typedef __attribute__((ext_vector_type(8)))  float    v8f;

union F16x16 { v16h v; uint4 q[2]; };
union H2Pack { _Float16 h2[2]; unsigned u; };

__device__ __forceinline__ v8f v8f_splat(float x) {
  v8f r;
#pragma unroll
  for (int e = 0; e < 8; ++e) r[e] = x;
  return r;
}

// ---------------- LDS layout (bytes), rows padded vs bank conflicts ----------------
// W1T rows: 128 -> 136 halves (272B = 68 dwords, lane stride 68%64=4 banks)
// W2T rows: 512 -> 520 halves; WattnT rows: 512 -> 516 floats (4-bank lane stride)
// x rows: 128 -> 136 halves; xpair rows: 32 -> 40 halves; x2 flat [512] f32
#define W1T_ROW   136
#define W2T_ROW   520
#define WAT_ROW   516
#define X_ROW     136
#define XP_ROW    40

#define OFF_W1T   0
#define SZ_W1T    (HID_ * W1T_ROW * 2)          // 139264
#define OFF_W2T   (OFF_W1T + SZ_W1T)
#define SZ_W2T    (BN_ * W2T_ROW * 2)           // 33280  -> 172544
#define OFF_WAT   (OFF_W2T + SZ_W2T)
#define SZ_WAT    (16 * WAT_ROW * 4)            // 33024  -> 205568
#define OFF_B1    (OFF_WAT + SZ_WAT)            // 2048   -> 207616
#define OFF_WP    (OFF_B1 + 2048)               // 512    -> 208128
#define OFF_BP    (OFF_WP + 512)                // 256    -> 208384
#define OFF_B2    (OFF_BP + 256)                // 128    -> 208512
#define OFF_BA    (OFF_B2 + 128)                // 64     -> 208576
#define OFF_WAVE  (OFF_BA + 64)
#define WAVE_STRIDE 7808
#define WOFF_X    0                              // 16*136 halves = 4352B
#define WOFF_XP   4352                           // 16*40 halves  = 1280B
#define WOFF_X2   5632                           // 512 floats    = 2048B
#define WOFF_WSM  7680                           // 16 floats
#define SMEM_TOTAL (OFF_WAVE + 4 * WAVE_STRIDE)  // 239808 < 320KB/WGP

// ================= kernel 1: weights -> transposed f16 =================
__global__ void prep_weights(const float* __restrict__ W1, const float* __restrict__ W2,
                             _Float16* __restrict__ W1T, _Float16* __restrict__ W2T) {
  int idx = blockIdx.x * 256 + threadIdx.x;      // 0 .. 81919
  if (idx < HID_ * 128) {                        // W1T[n][k] = W1[k][n]
    int n = idx >> 7, k = idx & 127;
    W1T[n * W1T_ROW + k] = (_Float16)W1[k * HID_ + n];
  } else {
    int i2 = idx - HID_ * 128;                   // W2T[n][k] = W2[k][n]
    if (i2 < BN_ * HID_) {
      int n = i2 >> 9, k = i2 & 511;
      W2T[n * W2T_ROW + k] = (_Float16)W2[k * BN_ + n];
    }
  }
}

// ================= kernel 2: neighbor ranking (1 wave / token) =================
// sel[m] = stable rank of pedestrian m by distance to i (reference's argsort(argsort) prefix)
__global__ void rank_kernel(const float* __restrict__ pos,
                            int* __restrict__ sel, float* __restrict__ relw) {
  __shared__ float d2s[4][224];
  const int wave = threadIdx.x >> 5, lane = threadIdx.x & 31;
  const int token = blockIdx.x * 4 + wave;
  const int s = token / N_, i = token - s * N_, base = s * N_;

  const float pix = pos[(base + i) * 2 + 0];
  const float piy = pos[(base + i) * 2 + 1];

  for (int j = lane; j < N_; j += 32) {
    float dx = pos[(base + j) * 2 + 0] - pix;
    float dy = pos[(base + j) * 2 + 1] - piy;
    d2s[wave][j] = dx * dx + dy * dy;
  }
  // same-wave LDS ops complete in order; stores above precede loads below
  const int m = lane >> 1, c = lane & 1;
  const float dm = d2s[wave][m];
  int cnt = 0;
#pragma unroll 4
  for (int j = c * 100; j < c * 100 + 100; ++j) {
    float dj = d2s[wave][j];
    cnt += (dj < dm) || (dj == dm && j < m);     // stable tie-break
  }
  cnt += __shfl_xor(cnt, 1, 32);
  if (c == 0) {
    sel[token * MAXP + m] = cnt;
    float qx = pos[(base + cnt) * 2 + 0];
    float qy = pos[(base + cnt) * 2 + 1];
    relw[token * 2 * MAXP + 2 * m + 0] = qx - pix;
    relw[token * 2 * MAXP + 2 * m + 1] = qy - piy;
  }
}

// ---- one layer-1 N-tile: preload 4 B frags (distinct regs), 4 WMMAs, relu, bounce ----
__device__ __forceinline__ void l1_tile(const _Float16* __restrict__ W1T_s,
                                        _Float16* __restrict__ xp_s,
                                        const F16x16* A, const float* __restrict__ b1_s,
                                        int t, int parity, int laneN, int half) {
  const _Float16* base = W1T_s + (16 * t + laneN) * W1T_ROW + 16 * half;
  F16x16 B0, B1, B2, B3;                     // all 8 ds_load_b128 in flight
  B0.q[0] = *(const uint4*)(base + 0);   B0.q[1] = *(const uint4*)(base + 8);
  B1.q[0] = *(const uint4*)(base + 32);  B1.q[1] = *(const uint4*)(base + 40);
  B2.q[0] = *(const uint4*)(base + 64);  B2.q[1] = *(const uint4*)(base + 72);
  B3.q[0] = *(const uint4*)(base + 96);  B3.q[1] = *(const uint4*)(base + 104);

  v8f acc = v8f_splat(b1_s[16 * t + laneN]);   // bias is per-N = per-lane in C layout
  acc = __builtin_amdgcn_wmma_f32_16x16x32_f16(false, A[0].v, false, B0.v, (short)0, acc, false, false);
  acc = __builtin_amdgcn_wmma_f32_16x16x32_f16(false, A[1].v, false, B1.v, (short)0, acc, false, false);
  acc = __builtin_amdgcn_wmma_f32_16x16x32_f16(false, A[2].v, false, B2.v, (short)0, acc, false, false);
  acc = __builtin_amdgcn_wmma_f32_16x16x32_f16(false, A[3].v, false, B3.v, (short)0, acc, false, false);

  // relu + bounce D layout -> layer-2 A layout through 1KB LDS pair buffer
#pragma unroll
  for (int r = 0; r < 8; ++r) {
    float v = acc[r]; v = v > 0.f ? v : 0.f;
    xp_s[(r + 8 * half) * XP_ROW + 16 * parity + laneN] = (_Float16)v;
  }
}

// ================= kernel 3: fused MLP + attention pool (WMMA) =================
__global__ void __launch_bounds__(128) attn_main(
    const float* __restrict__ h,      const float* __restrict__ Wpos,
    const float* __restrict__ bpos,   const float* __restrict__ b1,
    const float* __restrict__ b2,     const float* __restrict__ Wattn,
    const float* __restrict__ battn,
    const _Float16* __restrict__ W1Tg, const _Float16* __restrict__ W2Tg,
    const int* __restrict__ sel,      const float* __restrict__ relw,
    float* __restrict__ out) {
  __shared__ __align__(128) unsigned char smem[SMEM_TOTAL];

  _Float16* W1T_s = (_Float16*)(smem + OFF_W1T);
  _Float16* W2T_s = (_Float16*)(smem + OFF_W2T);
  float*    WaT_s = (float*)   (smem + OFF_WAT);
  float*    b1_s  = (float*)   (smem + OFF_B1);
  float*    wp_s  = (float*)   (smem + OFF_WP);
  float*    bp_s  = (float*)   (smem + OFF_BP);
  float*    b2_s  = (float*)   (smem + OFF_B2);
  float*    ba_s  = (float*)   (smem + OFF_BA);

  const int tid = threadIdx.x;
  // ---- stage all weights into LDS once per workgroup ----
  {
    const uint4* g = (const uint4*)W1Tg;  uint4* l = (uint4*)W1T_s;
    for (int k = tid; k < SZ_W1T / 16; k += 128) l[k] = g[k];
    g = (const uint4*)W2Tg;  l = (uint4*)W2T_s;
    for (int k = tid; k < SZ_W2T / 16; k += 128) l[k] = g[k];
    for (int k = tid; k < 512 * 16; k += 128) {  // transpose W_attn: WaT[j][p]
      int p = k >> 4, j = k & 15;
      WaT_s[j * WAT_ROW + p] = Wattn[k];
    }
    for (int k = tid; k < HID_; k += 128) b1_s[k] = b1[k];
    if (tid < 128) wp_s[tid] = Wpos[tid];
    if (tid < 64)  bp_s[tid] = bpos[tid];
    if (tid < 32)  b2_s[tid] = b2[tid];
    if (tid < 16)  ba_s[tid] = battn[tid];
  }
  __syncthreads();

  const int wave = tid >> 5, lane = tid & 31;
  const int laneN = lane & 15, half = lane >> 4;
  unsigned char* wbase = smem + OFF_WAVE + wave * WAVE_STRIDE;
  _Float16* x_s  = (_Float16*)(wbase + WOFF_X);
  _Float16* xp_s = (_Float16*)(wbase + WOFF_XP);
  float*    x2_s = (float*)   (wbase + WOFF_X2);   // flat [512]
  float*    wsm  = (float*)   (wbase + WOFF_WSM);

  const int gw = blockIdx.x * 4 + wave;            // 1000 waves -> 50 tokens each
  for (int token = gw; token < TOK; token += 1000) {
    const int s = token / N_;

    // prefetch next token's gather metadata (gfx1250 global_prefetch path)
    const int nt = token + 1000;
    if (nt < TOK) {
      __builtin_prefetch(sel + nt * MAXP, 0, 1);
      __builtin_prefetch(relw + nt * 2 * MAXP, 0, 1);
    }

    // ---- build x[16][128] = [h_sel | pos_embedding] in f16 (packed b32 stores) ----
    for (int m = 0; m < MAXP; ++m) {
      int j = sel[token * MAXP + m];
      const float* hr = h + (size_t)(s * N_ + j) * H_;
      float2 hv = *(const float2*)(hr + 2 * lane);
      H2Pack ph; ph.h2[0] = (_Float16)hv.x; ph.h2[1] = (_Float16)hv.y;
      *(unsigned*)(x_s + m * X_ROW + 2 * lane) = ph.u;

      float rx = relw[token * 2 * MAXP + 2 * m + 0];
      float ry = relw[token * 2 * MAXP + 2 * m + 1];
      int e0 = lane * 2;
      float v0 = fmaf(rx, wp_s[e0],     fmaf(ry, wp_s[64 + e0],     bp_s[e0]));
      float v1 = fmaf(rx, wp_s[e0 + 1], fmaf(ry, wp_s[64 + e0 + 1], bp_s[e0 + 1]));
      H2Pack pe; pe.h2[0] = (_Float16)v0; pe.h2[1] = (_Float16)v1;
      *(unsigned*)(x_s + m * X_ROW + 64 + e0) = pe.u;
    }

    // ---- A fragments for layer 1: 4 K-tiles of [16,32] f16 ----
    F16x16 A[4];
#pragma unroll
    for (int k = 0; k < 4; ++k) {
      const _Float16* p = x_s + laneN * X_ROW + 32 * k + 16 * half;
      A[k].q[0] = *(const uint4*)p;
      A[k].q[1] = *(const uint4*)(p + 8);
    }

    v8f acc2[2];  acc2[0] = v8f_splat(0.f);  acc2[1] = v8f_splat(0.f);

    // ---- layer 1 pairs fused into layer 2 accumulation ----
    for (int kt = 0; kt < 16; ++kt) {
      l1_tile(W1T_s, xp_s, A, b1_s, 2 * kt + 0, 0, laneN, half);
      l1_tile(W1T_s, xp_s, A, b1_s, 2 * kt + 1, 1, laneN, half);

      // layer-2 K-tile from the freshly bounced pair (in-wave DS ordering)
      F16x16 A2, B20, B21;
      const _Float16* pa = xp_s + laneN * XP_ROW + 16 * half;
      A2.q[0] = *(const uint4*)pa;
      A2.q[1] = *(const uint4*)(pa + 8);
      const _Float16* pb = W2T_s + laneN * W2T_ROW + 32 * kt + 16 * half;
      B20.q[0] = *(const uint4*)pb;
      B20.q[1] = *(const uint4*)(pb + 8);
      B21.q[0] = *(const uint4*)(pb + 16 * W2T_ROW);
      B21.q[1] = *(const uint4*)(pb + 16 * W2T_ROW + 8);
      acc2[0] = __builtin_amdgcn_wmma_f32_16x16x32_f16(false, A2.v, false, B20.v,
                                                       (short)0, acc2[0], false, false);
      acc2[1] = __builtin_amdgcn_wmma_f32_16x16x32_f16(false, A2.v, false, B21.v,
                                                       (short)0, acc2[1], false, false);
    }

    // ---- bias + relu on x2[16][32], stage flat to LDS (flat idx = m*32+c) ----
#pragma unroll
    for (int n2 = 0; n2 < 2; ++n2) {
      float bb = b2_s[16 * n2 + laneN];
#pragma unroll
      for (int r = 0; r < 8; ++r) {
        float v = acc2[n2][r] + bb; v = v > 0.f ? v : 0.f;
        x2_s[(r + 8 * half) * 32 + 16 * n2 + laneN] = v;
      }
    }

    // ---- logits: flat(x2)[512] @ W_attn[512,16]; lane=(j,half-of-K), float4 ----
    float part = 0.f;
    {
      const float* wrow = WaT_s + laneN * WAT_ROW + half * 256;
      const float* x2f  = x2_s + half * 256;
#pragma unroll 4
      for (int p = 0; p < 256; p += 4) {
        float4 wv = *(const float4*)(wrow + p);
        float4 xv = *(const float4*)(x2f + p);
        part = fmaf(xv.x, wv.x, part);
        part = fmaf(xv.y, wv.y, part);
        part = fmaf(xv.z, wv.z, part);
        part = fmaf(xv.w, wv.w, part);
      }
    }
    part += __shfl_xor(part, 16, 32);
    float logit = part + ba_s[laneN];

    // softmax over 16 within each half-wave (values duplicated across halves)
    float mx = logit;
#pragma unroll
    for (int d = 1; d < 16; d <<= 1) mx = fmaxf(mx, __shfl_xor(mx, d, 16));
    float ex = __expf(logit - mx);
    float sm = ex;
#pragma unroll
    for (int d = 1; d < 16; d <<= 1) sm += __shfl_xor(sm, d, 16);
    float wgt = ex / sm;
    if (lane < 16) wsm[lane] = wgt;

    // ---- weighted pool: out[c] = sum_m w[m]*x2[m][c], lane = c (stride-1, no conflicts) ----
    float o = 0.f;
#pragma unroll
    for (int m = 0; m < MAXP; ++m) o = fmaf(wsm[m], x2_s[m * 32 + lane], o);
    out[(size_t)token * BN_ + lane] = o;
  }
}

// ================= host launcher =================
extern "C" void kernel_launch(void* const* d_in, const int* in_sizes, int n_in,
                              void* d_out, int out_size, void* d_ws, size_t ws_size,
                              hipStream_t stream) {
  (void)in_sizes; (void)n_in; (void)out_size; (void)ws_size;
  const float* h_states = (const float*)d_in[0];
  const float* last_pos = (const float*)d_in[1];
  const float* W_pos    = (const float*)d_in[2];
  const float* b_pos    = (const float*)d_in[3];
  const float* W1       = (const float*)d_in[4];
  const float* b1       = (const float*)d_in[5];
  const float* W2       = (const float*)d_in[6];
  const float* b2       = (const float*)d_in[7];
  const float* W_attn   = (const float*)d_in[8];
  const float* b_attn   = (const float*)d_in[9];
  float* out = (float*)d_out;

  // workspace carve
  char* ws = (char*)d_ws;
  _Float16* W1T  = (_Float16*)ws;                               // 139264 B
  _Float16* W2T  = (_Float16*)(ws + SZ_W1T);                    // 33280 B
  int*      sel  = (int*)     (ws + SZ_W1T + SZ_W2T);           // 3.2 MB
  float*    relw = (float*)   (ws + SZ_W1T + SZ_W2T + (size_t)TOK * MAXP * 4); // 6.4 MB

  prep_weights<<<320, 256, 0, stream>>>(W1, W2, W1T, W2T);
  rank_kernel<<<TOK / 4, 128, 0, stream>>>(last_pos, sel, relw);
  attn_main<<<S_, 128, 0, stream>>>(h_states, W_pos, b_pos, b1, b2, W_attn, b_attn,
                                    W1T, W2T, sel, relw, out);
}
